// SLSTM_24094766531024
// MI455X (gfx1250) — compile-verified, matrix operand order checked by reading, not run on hardware
//
#include <hip/hip_runtime.h>

#define Bx 256
#define Sx 1024
#define Ix 64
#define Hx 128
#define Gx 512   // 4*H

typedef __attribute__((ext_vector_type(16))) __bf16 v16bf;
typedef __attribute__((ext_vector_type(8)))  float  v8f;

// Pointer types for the gfx1250 async global->LDS builtin:
// arg0: global int4*, arg1: LDS int4*, arg2: imm offset, arg3: cpol
typedef int v4i __attribute__((vector_size(16)));
typedef __attribute__((address_space(1))) v4i* gv4i_p;
typedef __attribute__((address_space(3))) v4i* lv4i_p;

#if __has_builtin(__builtin_amdgcn_global_load_async_to_lds_b128)
#define HAVE_ASYNC_LDS 1
#else
#define HAVE_ASYNC_LDS 0
#endif

__device__ __forceinline__ float sigmoid_f(float x) { return 1.f / (1.f + __expf(-x)); }
__device__ __forceinline__ float tanh_f(float x)    { return 2.f / (1.f + __expf(-2.f * x)) - 1.f; }

// Copy one 16x64 f32 x-tile for timestep t into LDS (256 threads, 1 float4 each).
__device__ __forceinline__ void load_x_tile(const float* __restrict__ Xg, int b0, int t,
                                            float* dst, int tid) {
  const int row = tid >> 4;
  const int col = (tid & 15) << 2;
  const float* src = Xg + ((size_t)(b0 + row) * Sx + t) * Ix + col;
  float* d = dst + row * Ix + col;
#if HAVE_ASYNC_LDS
  __builtin_amdgcn_global_load_async_to_lds_b128((gv4i_p)src, (lv4i_p)d, 0, 0);
#else
  *(float4*)d = *(const float4*)src;
#endif
}

__device__ __forceinline__ void wait_async() {
#if HAVE_ASYNC_LDS
#if __has_builtin(__builtin_amdgcn_s_wait_asynccnt)
  __builtin_amdgcn_s_wait_asynccnt(0);
#else
  asm volatile("s_wait_asynccnt 0" ::: "memory");
#endif
#endif
}

__global__ __launch_bounds__(256) void slstm_fused(
    const float* __restrict__ X, const float* __restrict__ W, const float* __restrict__ U,
    const float* __restrict__ bias, const float* __restrict__ Wy,
    float* __restrict__ hs_out, float* __restrict__ ht_out, float* __restrict__ ct_out)
{
  __shared__ float  xbuf[2][16][Ix];   // 8 KB, double-buffered x tile (f32)
  __shared__ __bf16 hbuf[2][16][Hx];   // 8 KB, double-buffered h state (bf16)

  const int tid  = threadIdx.x;
  const int wave = tid >> 5;
  const int lane = tid & 31;
  const int half = lane >> 4;   // lanes 16-31 handle K+16 / M+8 halves
  const int nloc = lane & 15;
  const int b0   = blockIdx.x * 16;

  // ---- Load constant B fragments (W, U columns for this wave's 4 gate tiles) into VGPRs.
  // B layout (32x16 bf16): lane = half*16 + N, VGPR v holds packed {K=kb+2v, K=kb+2v+1}.
  v16bf Wf[4][2], Uf[4][4];
  float bia[4], wyv[4];
  #pragma unroll
  for (int gi = 0; gi < 4; ++gi) {
    const int n = gi * Hx + wave * 16 + nloc;  // column in 4H gate space
    bia[gi] = bias[n];
    wyv[gi] = Wy[n];
    #pragma unroll
    for (int kk = 0; kk < 2; ++kk) {           // K = 64 over x@W
      const int kb = kk * 32 + half * 16;
      v16bf f;
      #pragma unroll
      for (int v = 0; v < 8; ++v) {
        f[2 * v]     = (__bf16)W[(size_t)(kb + 2 * v)     * Gx + n];
        f[2 * v + 1] = (__bf16)W[(size_t)(kb + 2 * v + 1) * Gx + n];
      }
      Wf[gi][kk] = f;
    }
    #pragma unroll
    for (int kk = 0; kk < 4; ++kk) {           // K = 128 over h@U
      const int kb = kk * 32 + half * 16;
      v16bf f;
      #pragma unroll
      for (int v = 0; v < 8; ++v) {
        f[2 * v]     = (__bf16)U[(size_t)(kb + 2 * v)     * Gx + n];
        f[2 * v + 1] = (__bf16)U[(size_t)(kb + 2 * v + 1) * Gx + n];
      }
      Uf[gi][kk] = f;
    }
  }

  // h(0) = 0
  for (int idx = tid; idx < 16 * Hx; idx += 256)
    ((__bf16*)hbuf[0])[idx] = (__bf16)0.f;

  load_x_tile(X, b0, 0, &xbuf[0][0][0], tid);
  wait_async();
  __syncthreads();

  v8f cst = {};                       // c-state tile (16 rows x 16 hidden cols), wave-local
  const int ncol = wave * 16 + nloc;  // hidden column owned by this lane

  for (int t = 0; t < Sx; ++t) {
    const int cur = t & 1, nxt = cur ^ 1;
    if (t + 1 < Sx) load_x_tile(X, b0, t + 1, &xbuf[nxt][0][0], tid);
    if ((t & 7) == 0 && t + 9 < Sx)
      __builtin_prefetch(X + ((size_t)(b0 + (tid >> 4)) * Sx + (t + 9)) * Ix, 0, 0);

    // ---- Build A fragments from LDS.
    // A layout (16x32 bf16): lane = half*16 + M, 16 contiguous K values starting at half*16.
    const int m  = nloc;
    const int kq = half * 16;
    v16bf ax[2], ah[4];
    #pragma unroll
    for (int kk = 0; kk < 2; ++kk) {
      const float* p = &xbuf[cur][m][kk * 32 + kq];
      v16bf f;
      #pragma unroll
      for (int v = 0; v < 16; ++v) f[v] = (__bf16)p[v];
      ax[kk] = f;
    }
    #pragma unroll
    for (int kk = 0; kk < 4; ++kk) {
      const __bf16* p = &hbuf[cur][m][kk * 32 + kq];
      v16bf f;
      #pragma unroll
      for (int v = 0; v < 16; ++v) f[v] = p[v];
      ah[kk] = f;
    }
    // y_t = X[:, t, 63] for the 8 C-layout rows M = r + 8*half
    float yv[8];
    #pragma unroll
    for (int r = 0; r < 8; ++r) yv[r] = xbuf[cur][r + 8 * half][Ix - 1];

    // ---- gates = bias + y*Wy + x@W + h@U   (24 WMMAs per wave per step)
    v8f gate[4];
    #pragma unroll
    for (int gi = 0; gi < 4; ++gi) {
      v8f acc;
      #pragma unroll
      for (int r = 0; r < 8; ++r) acc[r] = bia[gi] + yv[r] * wyv[gi];
      acc = __builtin_amdgcn_wmma_f32_16x16x32_bf16(false, ax[0], false, Wf[gi][0], (short)0, acc, false, false);
      acc = __builtin_amdgcn_wmma_f32_16x16x32_bf16(false, ax[1], false, Wf[gi][1], (short)0, acc, false, false);
      #pragma unroll
      for (int kk = 0; kk < 4; ++kk)
        acc = __builtin_amdgcn_wmma_f32_16x16x32_bf16(false, ah[kk], false, Uf[gi][kk], (short)0, acc, false, false);
      gate[gi] = acc;
    }

    // ---- LSTM cell update (f32 VALU, wave-local)
    v8f hnew;
    #pragma unroll
    for (int r = 0; r < 8; ++r) {
      const float it = sigmoid_f(gate[0][r]);
      const float ft = sigmoid_f(gate[1][r]);
      const float gt = tanh_f(gate[2][r]);
      const float ot = sigmoid_f(gate[3][r]);
      const float cc = ft * cst[r] + it * gt;
      cst[r]  = cc;
      hnew[r] = ot * tanh_f(cc);
    }

    // ---- Write h: bf16 copy to LDS for next step's A matrix, f32 to hidden_seq.
    #pragma unroll
    for (int r = 0; r < 8; ++r) {
      const int M = r + 8 * half;
      hbuf[nxt][M][ncol] = (__bf16)hnew[r];
      hs_out[((size_t)(b0 + M) * Sx + t) * Hx + ncol] = hnew[r];
    }
    if (t == Sx - 1) {
      #pragma unroll
      for (int r = 0; r < 8; ++r) {
        const int M = r + 8 * half;
        ht_out[(size_t)(b0 + M) * Hx + ncol] = hnew[r];
        ct_out[(size_t)(b0 + M) * Hx + ncol] = cst[r];
      }
    }
    wait_async();
    __syncthreads();
  }
}

// y_pred = h_T @ fc_w + fc_b  (256x128 @ 128x1, trivial)
__global__ void slstm_ypred(const float* __restrict__ ht, const float* __restrict__ fc_w,
                            const float* __restrict__ fc_b, float* __restrict__ y_pred)
{
  const int b = blockIdx.x * blockDim.x + threadIdx.x;
  if (b < Bx) {
    float s = fc_b[0];
    const float* h = ht + (size_t)b * Hx;
    #pragma unroll 8
    for (int k = 0; k < Hx; ++k) s += h[k] * fc_w[k];
    y_pred[b] = s;
  }
}

extern "C" void kernel_launch(void* const* d_in, const int* in_sizes, int n_in,
                              void* d_out, int out_size, void* d_ws, size_t ws_size,
                              hipStream_t stream) {
  const float* X    = (const float*)d_in[0];
  const float* W    = (const float*)d_in[1];
  const float* U    = (const float*)d_in[2];
  const float* bias = (const float*)d_in[3];
  const float* Wy   = (const float*)d_in[4];
  const float* fc_w = (const float*)d_in[5];
  const float* fc_b = (const float*)d_in[6];

  float* out    = (float*)d_out;
  float* y_pred = out;                          // (256,1)
  float* hs     = out + Bx;                     // (256,1024,128)
  float* ht     = hs + (size_t)Bx * Sx * Hx;    // (256,128)
  float* ct     = ht + (size_t)Bx * Hx;         // (256,128)

  slstm_fused<<<Bx / 16, 256, 0, stream>>>(X, W, U, bias, Wy, hs, ht, ct);
  slstm_ypred<<<1, 256, 0, stream>>>(ht, fc_w, fc_b, y_pred);
}
